// CustomerRiskGNN_44555990729321
// MI455X (gfx1250) — matrix operand order repaired and checked by reading, not compiled, not run on hardware
//
#include <hip/hip_runtime.h>
#include <hip/hip_bf16.h>

typedef __attribute__((ext_vector_type(16))) _Float16 v16h;
typedef __attribute__((ext_vector_type(8)))  float    v8f;

#define NN 50000
#define EE 800000
#define NEG_SLOPE 0.2f
#define BN_EPS 1e-5f

// ---------------- degree count ----------------
__global__ void count_edges(const long long* __restrict__ dst,
                            float* __restrict__ cnt, int nE) {
    int e = blockIdx.x * blockDim.x + threadIdx.x;
    if (e < nE) atomicAdd(&cnt[(int)dst[e]], 1.0f);
}

// ---------------- neighbor scatter-add ----------------
// one thread per (edge, 4-float chunk): float4 gather + 4 float atomics
__global__ void scatter_add(const float* __restrict__ feat,
                            const long long* __restrict__ src,
                            const long long* __restrict__ dst,
                            float* __restrict__ agg, int nE) {
    long long tid = (long long)blockIdx.x * blockDim.x + threadIdx.x;
    if (tid >= (long long)nE * 16) return;
    int e = (int)(tid >> 4);
    int c = ((int)tid & 15) * 4;
    int s = (int)src[e];
    int d = (int)dst[e];
    float4 v = *(const float4*)(feat + (long long)s * 64 + c);
    float* ap = agg + (long long)d * 64 + c;
    atomicAdd(ap + 0, v.x);
    atomicAdd(ap + 1, v.y);
    atomicAdd(ap + 2, v.z);
    atomicAdd(ap + 3, v.w);
}

// ---------------- fused conv: h=(x+agg)/(1+cnt); y=leaky(h@W+b); z=leaky(BN(y)) ----------------
// 128 threads = 4 waves/block; each wave: one 16-row slab x all 64 output cols.
// 8x v_wmma_f32_16x16x32_f16 per wave (4 N-tiles * 2 K-steps of 32).
__global__ __launch_bounds__(128) void conv_wmma(
    const float* __restrict__ xin, const float* __restrict__ agg,
    const float* __restrict__ cnt, const float* __restrict__ W,
    const float* __restrict__ bias,
    const float* __restrict__ g,  const float* __restrict__ beta,
    const float* __restrict__ rm, const float* __restrict__ rv,
    float* __restrict__ out, int n_nodes) {
    __shared__ _Float16 Ws[64 * 64];  // W[k][n] as f16, 8 KB
    for (int i = threadIdx.x; i < 64 * 64; i += 128)
        Ws[i] = (_Float16)W[i];
    __syncthreads();

    const int wave = threadIdx.x >> 5;
    const int lane = threadIdx.x & 31;
    const int slab = blockIdx.x * 4 + wave;
    const int row0 = slab * 16;
    if (row0 >= n_nodes) return;  // wave-uniform: EXEC stays all-ones for WMMA

    // ---- A fragments (16x32 f16, two K-blocks). contiguous K order used here (B matches):
    // lane L: M = L&15 ; half j -> K = j + 16*(L>>4)
    const int M   = row0 + (lane & 15);
    const int hk  = lane >> 4;
    const float inv = 1.0f / (1.0f + fmaxf(cnt[M], 1.0f));
    const float* xr = xin + (long long)M * 64;
    const float* ar = agg + (long long)M * 64;
    v16h a0, a1;
#pragma unroll
    for (int j = 0; j < 16; j += 2) {
        int K = j + 16 * hk;
        float2 xv = *(const float2*)(xr + K);
        float2 av = *(const float2*)(ar + K);
        a0[j]     = (_Float16)((xv.x + av.x) * inv);
        a0[j + 1] = (_Float16)((xv.y + av.y) * inv);
        float2 xw = *(const float2*)(xr + K + 32);
        float2 aw = *(const float2*)(ar + K + 32);
        a1[j]     = (_Float16)((xw.x + aw.x) * inv);
        a1[j + 1] = (_Float16)((xw.y + aw.y) * inv);
    }

    // ---- 4 output column tiles of 16
#pragma unroll
    for (int t = 0; t < 4; ++t) {
        const int n  = t * 16 + (lane & 15);
        const int kb = hk * 16;  // B layout: lanes 0-15 K=0..15, lanes 16-31 K=16..31
        v16h b0, b1;
#pragma unroll
        for (int j = 0; j < 16; ++j) {
            b0[j] = Ws[(kb + j) * 64 + n];
            b1[j] = Ws[(32 + kb + j) * 64 + n];
        }
        v8f acc = {};
        acc = __builtin_amdgcn_wmma_f32_16x16x32_f16(false, a0, false, b0,
                                                     (short)0, acc, false, false);
        acc = __builtin_amdgcn_wmma_f32_16x16x32_f16(false, a1, false, b1,
                                                     (short)0, acc, false, false);

        // fused epilogue: +bias, leaky, BN affine, leaky
        const float bn_s = g[n] * rsqrtf(rv[n] + BN_EPS);
        const float bn_b = beta[n] - rm[n] * bn_s;
        const float bv   = bias[n];
#pragma unroll
        for (int r = 0; r < 8; ++r) {
            int Ml = hk * 8 + r;  // C/D layout: VGPR r -> M = r + 8*(lane>>4)
            float v = acc[r] + bv;
            v = (v > 0.0f) ? v : NEG_SLOPE * v;
            v = v * bn_s + bn_b;
            v = (v > 0.0f) ? v : NEG_SLOPE * v;
            out[(long long)(row0 + Ml) * 64 + n] = v;
        }
    }
}

// ---------------- fused classifier: relu(h@Wc1+bc1)@Wc2+bc2 ----------------
__global__ __launch_bounds__(256) void classifier(
    const float* __restrict__ h, const float* __restrict__ Wc1,
    const float* __restrict__ bc1, const float* __restrict__ Wc2,
    const float* __restrict__ bc2, float* __restrict__ out, int n_nodes) {
    __shared__ float W1s[64 * 32];
    __shared__ float W2s[32 * 2];
    __shared__ float b1s[32];
    __shared__ float b2s[2];
    for (int i = threadIdx.x; i < 64 * 32; i += 256) W1s[i] = Wc1[i];
    if (threadIdx.x < 64) W2s[threadIdx.x] = Wc2[threadIdx.x];
    if (threadIdx.x < 32) b1s[threadIdx.x] = bc1[threadIdx.x];
    if (threadIdx.x < 2)  b2s[threadIdx.x] = bc2[threadIdx.x];
    __syncthreads();

    int node = blockIdx.x * 256 + threadIdx.x;
    if (node >= n_nodes) return;

    float hv[64];
    const float4* hp = (const float4*)(h + (long long)node * 64);
#pragma unroll
    for (int i = 0; i < 16; ++i) {
        float4 v = hp[i];
        hv[4 * i]     = v.x;
        hv[4 * i + 1] = v.y;
        hv[4 * i + 2] = v.z;
        hv[4 * i + 3] = v.w;
    }
    float o0 = b2s[0], o1 = b2s[1];
#pragma unroll 4
    for (int j = 0; j < 32; ++j) {
        float a = b1s[j];
#pragma unroll
        for (int k = 0; k < 64; ++k) a = fmaf(hv[k], W1s[k * 32 + j], a);
        a = fmaxf(a, 0.0f);
        o0 = fmaf(a, W2s[j * 2 + 0], o0);
        o1 = fmaf(a, W2s[j * 2 + 1], o1);
    }
    out[(long long)node * 2 + 0] = o0;
    out[(long long)node * 2 + 1] = o1;
}

extern "C" void kernel_launch(void* const* d_in, const int* in_sizes, int n_in,
                              void* d_out, int out_size, void* d_ws, size_t ws_size,
                              hipStream_t stream) {
    const int N = in_sizes[0] / 64;   // 50000
    const int E = in_sizes[1] / 2;    // 800000

    const float*     x    = (const float*)d_in[0];
    const long long* ei   = (const long long*)d_in[1];  // int64 [2, E]
    const long long* src  = ei;
    const long long* dst  = ei + E;
    const float* W1   = (const float*)d_in[2];
    const float* b1   = (const float*)d_in[3];
    const float* W2   = (const float*)d_in[4];
    const float* b2   = (const float*)d_in[5];
    const float* g1   = (const float*)d_in[6];
    const float* bet1 = (const float*)d_in[7];
    const float* rm1  = (const float*)d_in[8];
    const float* rv1  = (const float*)d_in[9];
    const float* g2   = (const float*)d_in[10];
    const float* bet2 = (const float*)d_in[11];
    const float* rm2  = (const float*)d_in[12];
    const float* rv2  = (const float*)d_in[13];
    const float* Wc1  = (const float*)d_in[14];
    const float* bc1  = (const float*)d_in[15];
    const float* Wc2  = (const float*)d_in[16];
    const float* bc2  = (const float*)d_in[17];
    float* out = (float*)d_out;

    // workspace layout
    float* agg = (float*)d_ws;           // N*64
    float* cnt = agg + (size_t)N * 64;   // N
    float* h1  = cnt + N;                // N*64
    float* h2  = h1 + (size_t)N * 64;    // N*64

    const int slabs      = (N + 15) / 16;
    const int conv_grid  = (slabs + 3) / 4;
    const long long scat = (long long)E * 16;
    const int scat_grid  = (int)((scat + 255) / 256);

    // --- layer 1 ---
    hipMemsetAsync(agg, 0, (size_t)N * 64 * sizeof(float), stream);
    hipMemsetAsync(cnt, 0, (size_t)N * sizeof(float), stream);
    count_edges<<<(E + 255) / 256, 256, 0, stream>>>(dst, cnt, E);
    scatter_add<<<scat_grid, 256, 0, stream>>>(x, src, dst, agg, E);
    conv_wmma<<<conv_grid, 128, 0, stream>>>(x, agg, cnt, W1, b1,
                                             g1, bet1, rm1, rv1, h1, N);
    // --- layer 2 ---
    hipMemsetAsync(agg, 0, (size_t)N * 64 * sizeof(float), stream);
    scatter_add<<<scat_grid, 256, 0, stream>>>(h1, src, dst, agg, E);
    conv_wmma<<<conv_grid, 128, 0, stream>>>(h1, agg, cnt, W2, b2,
                                             g2, bet2, rm2, rv2, h2, N);
    // --- head ---
    classifier<<<(N + 255) / 256, 256, 0, stream>>>(h2, Wc1, bc1, Wc2, bc2, out, N);
}
